// Retrieval_Retro_8211977470578
// MI455X (gfx1250) — compile-verified, matrix operand order checked by reading, not run on hardware
//
#include <hip/hip_runtime.h>
#include <stdint.h>

// ---------------- types ----------------
typedef __attribute__((ext_vector_type(16))) __bf16          v16bf;
typedef __attribute__((ext_vector_type(16))) unsigned short  v16u;
typedef __attribute__((ext_vector_type(8)))  unsigned short  v8u;
typedef __attribute__((ext_vector_type(4)))  unsigned short  v4u;
typedef __attribute__((ext_vector_type(4)))  float           v4f;
typedef __attribute__((ext_vector_type(8)))  float           v8f;
typedef __attribute__((ext_vector_type(4)))  int             v4i;

static __device__ __forceinline__ unsigned short f2bf(float f) {
  union { float f; unsigned u; } x; x.f = f;
  unsigned u = x.u + 0x7FFFu + ((x.u >> 16) & 1u);   // round-to-nearest-even
  return (unsigned short)(u >> 16);
}
static __device__ __forceinline__ float bf2f(unsigned short h) {
  union { unsigned u; float f; } x; x.u = ((unsigned)h) << 16;
  return x.f;
}

// ---- CDNA5 async global->LDS copy (GLOBAL_LOAD_ASYNC_TO_LDS_B128, ASYNCcnt) ----
#if defined(__has_builtin)
#if __has_builtin(__builtin_amdgcn_global_load_async_to_lds_b128) && \
    __has_builtin(__builtin_amdgcn_s_wait_asynccnt)
#define HAVE_ASYNC_LDS 1
#endif
#endif

#ifdef HAVE_ASYNC_LDS
typedef __attribute__((address_space(1))) v4i* as1_v4i_p;   // global ("__device__")
typedef __attribute__((address_space(3))) v4i* as3_v4i_p;   // LDS ("__shared__")
static __device__ __forceinline__ void async_cp16(const unsigned short* g, unsigned short* l) {
  __builtin_amdgcn_global_load_async_to_lds_b128(
      (as1_v4i_p)g, (as3_v4i_p)l, 0, 0);
}
static __device__ __forceinline__ void async_wait() { __builtin_amdgcn_s_wait_asynccnt(0); }
#else
static __device__ __forceinline__ void async_cp16(const unsigned short* g, unsigned short* l) {
  *(v8u*)l = *(const v8u*)g;
}
static __device__ __forceinline__ void async_wait() {}
#endif

// ---------------- generic WMMA GEMM (template-specialized) ----------------
// C[M,N] = epilogue( A[M,K] @ W[K,N] + bias ),  A = concat(src0[, src1]) along K.
// S1:   0 = no src1, 1 = src1 f32, 2 = src1 bf16
// GATHER: row-gather src0 through gidx
// A0F:  src0 is f32 (else bf16)
// OUTM: 0 = store bf16, 1 = store f32, 2 = atomic f32 scatter (fused segment_sum)
// Assumes: M % 128 == 0, N % 128 == 0, all K region boundaries % 8 == 0.
// Block tile 128x128, 8 wave32 waves; each wave: 2 A-frags x 4 B-frags = 8 WMMA / K-step.
// Double-buffered LDS; A staged via async global->LDS when source is bf16.
#define BM 128
#define BN 128
#define BK 32

template<int S1, bool GATHER, bool A0F, int OUTM>
__global__ __launch_bounds__(256) void gemm_wmma(
    const void* __restrict__ A0, int ld0, int k0len,
    const void* __restrict__ A1, int ld1, int k1len,
    const int*  __restrict__ gidx,
    const unsigned short* __restrict__ W, int ldw,
    const float* __restrict__ bias,
    int K,
    int act, const float* __restrict__ alphaPtr,   // 0 none,1 relu,2 prelu,3 sigmoid
    const unsigned short* __restrict__ residual, int ldr,
    void* __restrict__ Cout, int ldc,
    float* __restrict__ scatOut, const int* __restrict__ scatIdx, int ldscat)
{
  __shared__ __align__(16) unsigned short sA[2][BM * BK];       // [row][k]
  __shared__ __align__(16) unsigned short sB[2][8 * 32 * 16];   // [s][lane][j] frag-swizzled

  const int t    = threadIdx.x;
  const int lane = t & 31;
  const int wave = t >> 5;
  const int mw   = wave & 3;   // M position: rows [mw*32, mw*32+32)
  const int ng   = wave >> 2;  // N group 0..1: cols [ng*64, ng*64+64)
  const int rowBase = blockIdx.y * BM;
  const int nBase   = blockIdx.x * BN;

  // ---- A staging: thread -> rows {r0, r0+64}, one 8-wide k chunk each (same col)
  const int acol = (t & 3) * 8;
  const int r0   = t >> 2;                 // 0..63
  const char* a0b[2];
  const char* a1b[2] = {nullptr, nullptr};
  {
    const long esz0 = A0F ? 4 : 2;
    const long gr0 = rowBase + r0, gr1 = rowBase + r0 + 64;
    const long s0r0 = GATHER ? (long)gidx[gr0] : gr0;
    const long s0r1 = GATHER ? (long)gidx[gr1] : gr1;
    a0b[0] = (const char*)A0 + s0r0 * ld0 * esz0;
    a0b[1] = (const char*)A0 + s0r1 * ld0 * esz0;
    if (S1) {
      const long esz1 = (S1 == 1) ? 4 : 2;
      a1b[0] = (const char*)A1 + gr0 * ld1 * esz1;
      a1b[1] = (const char*)A1 + gr1 * ld1 * esz1;
    }
  }

  auto stageA = [&](int kt, int buf) {
    const int kk = kt * BK + acol;
    #pragma unroll
    for (int j = 0; j < 2; ++j) {
      unsigned short* dst = &sA[buf][(r0 + j * 64) * BK + acol];
      if (kk < k0len) {
        if (A0F) {
          const float* p = (const float*)(a0b[j] + (long)kk * 4);
          v4f lo = *(const v4f*)p, hi = *(const v4f*)(p + 4);
          v8u out;
          #pragma unroll
          for (int i = 0; i < 4; ++i) { out[i] = f2bf(lo[i]); out[i + 4] = f2bf(hi[i]); }
          *(v8u*)dst = out;
        } else {
          async_cp16((const unsigned short*)(a0b[j] + (long)kk * 2), dst);
        }
      } else if (S1 && kk < k0len + k1len) {
        const int kk1 = kk - k0len;
        if (S1 == 1) {
          const float* p = (const float*)(a1b[j] + (long)kk1 * 4);
          v4f lo = *(const v4f*)p, hi = *(const v4f*)(p + 4);
          v8u out;
          #pragma unroll
          for (int i = 0; i < 4; ++i) { out[i] = f2bf(lo[i]); out[i + 4] = f2bf(hi[i]); }
          *(v8u*)dst = out;
        } else {
          async_cp16((const unsigned short*)(a1b[j] + (long)kk1 * 2), dst);
        }
      } else {
        v8u z = {0,0,0,0,0,0,0,0};
        *(v8u*)dst = z;
      }
    }
  };

  // ---- B staging: 4x4 in-register transpose -> 4x ds_store_b64
  const int cg = t & 3;            // col quad within subtile
  const int bs = (t >> 2) & 7;     // n subtile 0..7
  const int kq = t >> 5;           // k quad 0..7
  const int bhigh = (kq >> 1) & 1;
  const int j4    = (kq & 1) * 4 + ((kq >> 2) & 1) * 8;
  const int gcol  = nBase + bs * 16 + cg * 4;

  auto stageB = [&](int kt, int buf) {
    const int kw0 = kt * BK + kq * 4;
    const unsigned short* wp = W + (long)kw0 * ldw + gcol;
    v4u r[4];
    if (kw0 + 3 < K) {                       // common: whole quad in-bounds
      #pragma unroll
      for (int rr = 0; rr < 4; ++rr) r[rr] = *(const v4u*)(wp + (long)rr * ldw);
    } else {                                 // K tail (embed only)
      #pragma unroll
      for (int rr = 0; rr < 4; ++rr) {
        v4u z = {0,0,0,0};
        r[rr] = (kw0 + rr < K) ? *(const v4u*)(wp + (long)rr * ldw) : z;
      }
    }
    #pragma unroll
    for (int c = 0; c < 4; ++c) {
      v4u o = { r[0][c], r[1][c], r[2][c], r[3][c] };
      const int lane_ = bhigh * 16 + cg * 4 + c;
      *(v4u*)&sB[buf][((bs * 32 + lane_) << 4) + j4] = o;
    }
  };

  v8f zero = {0.f,0.f,0.f,0.f,0.f,0.f,0.f,0.f};
  v8f acc[2][4] = {{zero, zero, zero, zero}, {zero, zero, zero, zero}};

  const int frBase = mw * 32 + (lane & 15);
  const int half   = lane >> 4;
  const int ksteps = (K + BK - 1) / BK;

  stageA(0, 0);
  stageB(0, 0);
  for (int kt = 0; kt < ksteps; ++kt) {
    const int buf = kt & 1;
    async_wait();            // make this thread's async LDS writes visible pre-barrier
    __syncthreads();
    if (kt + 1 < ksteps) {
      if (kt + 2 < ksteps)
        __builtin_prefetch(W + (long)((kt + 2) * BK + kq * 4) * ldw + gcol, 0, 0);
      stageA(kt + 1, buf ^ 1);
      stageB(kt + 1, buf ^ 1);
    }
    // two A fragments (rows mw*32 and mw*32+16)
    union AU { v16u v; v8u h[2]; } au0, au1;
    au0.h[0] = *(const v8u*)&sA[buf][frBase * BK + half * 8];
    au0.h[1] = *(const v8u*)&sA[buf][frBase * BK + 16 + half * 8];
    au1.h[0] = *(const v8u*)&sA[buf][(frBase + 16) * BK + half * 8];
    au1.h[1] = *(const v8u*)&sA[buf][(frBase + 16) * BK + 16 + half * 8];
    v16bf af0 = __builtin_bit_cast(v16bf, au0.v);
    v16bf af1 = __builtin_bit_cast(v16bf, au1.v);
    const v16u* bp = (const v16u*)sB[buf];
    #pragma unroll
    for (int q = 0; q < 4; ++q) {
      v16bf bfrag = __builtin_bit_cast(v16bf, bp[(ng * 4 + q) * 32 + lane]);
      acc[0][q] = __builtin_amdgcn_wmma_f32_16x16x32_bf16(
                      false, af0, false, bfrag, (short)0, acc[0][q], false, false);
      acc[1][q] = __builtin_amdgcn_wmma_f32_16x16x32_bf16(
                      false, af1, false, bfrag, (short)0, acc[1][q], false, false);
    }
  }

  // ---- epilogue
  const float alpha = (act == 2) ? alphaPtr[0] : 0.f;
  float bv[4];
  int   cols[4];
  #pragma unroll
  for (int q = 0; q < 4; ++q) {
    cols[q] = nBase + (ng * 4 + q) * 16 + (lane & 15);
    bv[q] = bias ? bias[cols[q]] : 0.f;
  }
  #pragma unroll
  for (int a = 0; a < 2; ++a) {
    #pragma unroll
    for (int i = 0; i < 8; ++i) {
      const int row = rowBase + mw * 32 + a * 16 + i + 8 * half;
      int d = 0;
      if (OUTM == 2) d = scatIdx[row];
      #pragma unroll
      for (int q = 0; q < 4; ++q) {
        float v = acc[a][q][i] + bv[q];
        if (act == 1)      v = v > 0.f ? v : 0.f;
        else if (act == 2) v = v >= 0.f ? v : alpha * v;
        else if (act == 3) v = 1.f / (1.f + __expf(-v));
        if (residual) v += bf2f(residual[(long)row * ldr + cols[q]]);
        if (OUTM == 2)      atomicAdd(&scatOut[(long)d * ldscat + cols[q]], v);
        else if (OUTM == 1) ((float*)Cout)[(long)row * ldc + cols[q]] = v;
        else                ((unsigned short*)Cout)[(long)row * ldc + cols[q]] = f2bf(v);
      }
    }
  }
}

// ---------------- layernorm over 256 cols (one block per row) ----------------
__global__ __launch_bounds__(256) void ln_kernel(
    const unsigned short* __restrict__ x,
    const float* __restrict__ g, const float* __restrict__ b,
    unsigned short* __restrict__ y)
{
  __shared__ float red[32];
  const int row = blockIdx.x, t = threadIdx.x;
  const float v = bf2f(x[(long)row * 256 + t]);
  float s = v, s2 = v * v;
  for (int o = 16; o; o >>= 1) { s += __shfl_down(s, o, 32); s2 += __shfl_down(s2, o, 32); }
  const int wave = t >> 5, lane = t & 31;
  if (lane == 0) { red[wave] = s; red[8 + wave] = s2; }
  __syncthreads();
  if (t == 0) {
    float ts = 0.f, ts2 = 0.f;
    for (int i = 0; i < 8; ++i) { ts += red[i]; ts2 += red[8 + i]; }
    red[16] = ts; red[17] = ts2;
  }
  __syncthreads();
  const float mean = red[16] * (1.f / 256.f);
  const float var  = red[17] * (1.f / 256.f) - mean * mean;
  const float inv  = rsqrtf(var + 1e-5f);
  y[(long)row * 256 + t] = f2bf((v - mean) * inv * g[t] + b[t]);
}

// ---------------- weighted segment pooling ----------------
__global__ __launch_bounds__(256) void pool_kernel(
    const unsigned short* __restrict__ x, const float* __restrict__ fw,
    const int* __restrict__ batch, float* __restrict__ emb, int Nn)
{
  const int n = blockIdx.x;
  if (n >= Nn) return;
  const int c = threadIdx.x;
  const float v = bf2f(x[(long)n * 256 + c]) * fw[n];
  atomicAdd(&emb[(long)batch[n] * 256 + c], v);
}

// ---------------- generic 2D cast/copy (f32<->bf16) ----------------
__global__ __launch_bounds__(256) void cast2d(
    const void* __restrict__ src, int srcF32, int sld,
    void* __restrict__ dst, int dstF32, int dld, long Mr, long Cc)
{
  const long i = (long)blockIdx.x * blockDim.x + threadIdx.x;
  if (i >= Mr * Cc) return;
  const long r = i / Cc, c = i % Cc;
  const float v = srcF32 ? ((const float*)src)[r * sld + c]
                         : bf2f(((const unsigned short*)src)[r * sld + c]);
  if (dstF32) ((float*)dst)[r * dld + c] = v;
  else        ((unsigned short*)dst)[r * dld + c] = f2bf(v);
}

// ---------------- templated launcher ----------------
template<int S1, bool GATHER, bool A0F, int OUTM>
static void launch_gemm(hipStream_t stream,
    const void* A0, int ld0, int k0,
    const void* A1, int ld1, int k1,
    const int* gidx,
    const unsigned short* W, int ldw, const float* bias,
    int M, int N, int act, const float* alpha,
    const unsigned short* res, int ldr,
    void* C, int ldc,
    float* scat, const int* sidx, int ldscat)
{
  dim3 grid(N / BN, M / BM);
  gemm_wmma<S1, GATHER, A0F, OUTM><<<grid, 256, 0, stream>>>(
      A0, ld0, k0, A1, ld1, k1, gidx, W, ldw, bias, k0 + k1,
      act, alpha, res, ldr, C, ldc, scat, sidx, ldscat);
}

// ---------------- host orchestration ----------------
extern "C" void kernel_launch(void* const* d_in, const int* in_sizes, int n_in,
                              void* d_out, int out_size, void* d_ws, size_t ws_size,
                              hipStream_t stream)
{
  (void)in_sizes; (void)out_size; (void)ws_size;
  auto IN = [&](int i) { return d_in[(i < n_in && i >= 0) ? i : 0]; };

  // graph inputs (insertion order of setup_inputs dict)
  const float* gx[3]  = { (const float*)IN(0),  (const float*)IN(5),  (const float*)IN(10) };
  const int*   gei[3] = { (const int*)IN(1),    (const int*)IN(6),    (const int*)IN(11)  };
  const float* gea[3] = { (const float*)IN(2),  (const float*)IN(7),  (const float*)IN(12) };
  const float* gfw[3] = { (const float*)IN(3),  (const float*)IN(8),  (const float*)IN(13) };
  const int*   gbt[3] = { (const int*)IN(4),    (const int*)IN(9),    (const int*)IN(14)  };
  const int    gN[3]  = { 131072, 65536, 65536 };
  const int    gE[3]  = { 524288, 262144, 262144 };

  // params flattened: sorted pytree keys -> ca1,ca2,cls,fuse1,fuse2,gnn,sa1,sa2
  struct Enc { const float *ff1W,*ff1b,*ff2W,*ff2b,*ln1b,*ln1g,*ln2b,*ln2g,*outW,*outb,*qkvW,*qkvb; };
  auto getEnc = [&](int base) {
    Enc e;
    e.ff1W=(const float*)IN(base+0);  e.ff1b=(const float*)IN(base+1);
    e.ff2W=(const float*)IN(base+2);  e.ff2b=(const float*)IN(base+3);
    e.ln1b=(const float*)IN(base+4);  e.ln1g=(const float*)IN(base+5);
    e.ln2b=(const float*)IN(base+6);  e.ln2g=(const float*)IN(base+7);
    e.outW=(const float*)IN(base+8);  e.outb=(const float*)IN(base+9);
    e.qkvW=(const float*)IN(base+10); e.qkvb=(const float*)IN(base+11);
    return e;
  };
  Enc ca1 = getEnc(15), ca2 = getEnc(27), sa1 = getEnc(56), sa2 = getEnc(68);
  const float* clsW1 = (const float*)IN(39);
  const float* clsW2 = (const float*)IN(40);
  const float* clsA  = (const float*)IN(41);
  const float* clsB1 = (const float*)IN(42);
  const float* clsB2 = (const float*)IN(43);
  const float* fuseW[2] = { (const float*)IN(44), (const float*)IN(47) };
  const float* fuseA[2] = { (const float*)IN(45), (const float*)IN(48) };
  const float* fuseB[2] = { (const float*)IN(46), (const float*)IN(49) };
  const float* embW = (const float*)IN(50);
  const float* embB = (const float*)IN(51);
  const float* msgW = (const float*)IN(52);
  const float* msgB = (const float*)IN(53);
  const float* updW = (const float*)IN(54);
  const float* updB = (const float*)IN(55);

  // workspace bump allocator
  char* wsp = (char*)d_ws; size_t off = 0;
  auto alloc = [&](size_t bytes) -> void* {
    void* p = wsp + off; off += (bytes + 255) & ~(size_t)255; return p;
  };

  auto castTo = [&](const void* src,int sF32,int sld, void* dst,int dF32,int dld,long Mr,long Cc)
  {
    long tot = Mr * Cc;
    cast2d<<<(unsigned)((tot + 255) / 256), 256, 0, stream>>>(src,sF32,sld, dst,dF32,dld, Mr,Cc);
  };
  auto castW = [&](const float* src, long rows, long cols) -> unsigned short* {
    unsigned short* d = (unsigned short*)alloc((size_t)rows * cols * 2);
    castTo(src, 1, (int)cols, d, 0, (int)cols, rows, cols);
    return d;
  };

  // ---- bf16 weight copies (tiny vs GEMM cost)
  unsigned short* embWbf = castW(embW, 200, 256);
  unsigned short* msgWbf = castW(msgW, 3 * 384, 256);
  unsigned short* updWbf = castW(updW, 3 * 512, 256);
  unsigned short* fuseWbf[2] = { castW(fuseW[0], 512, 256), castW(fuseW[1], 512, 256) };
  struct EncBf { unsigned short *q,*o,*f1,*f2; };
  auto castEnc = [&](const Enc& e) {
    EncBf b;
    b.q  = castW(e.qkvW, 2 * 256, 768);
    b.o  = castW(e.outW, 2 * 256, 256);
    b.f1 = castW(e.ff1W, 2 * 256, 1024);
    b.f2 = castW(e.ff2W, 2 * 1024, 256);
    return b;
  };
  EncBf sa1b = castEnc(sa1), ca1b = castEnc(ca1), sa2b = castEnc(sa2), ca2b = castEnc(ca2);
  unsigned short* clsW1bf = castW(clsW1, 768, 768);
  unsigned short* clsW2bf = castW(clsW2, 768, 1024);

  // ---- activation workspace (sized for main graph)
  unsigned short* x_cur = (unsigned short*)alloc((size_t)131072 * 256 * 2);
  unsigned short* x_nxt = (unsigned short*)alloc((size_t)131072 * 256 * 2);
  float*          agg   = (float*)alloc((size_t)131072 * 256 * 4);
  float* embF[3] = { (float*)alloc(256*256*4), (float*)alloc(256*256*4), (float*)alloc(256*256*4) };
  unsigned short* hbuf  = (unsigned short*)alloc(256 * 256 * 2);
  unsigned short* vbuf  = (unsigned short*)alloc(256 * 256 * 2);
  unsigned short* fbuf  = (unsigned short*)alloc(256 * 1024 * 2);
  unsigned short* sx    = (unsigned short*)alloc(256 * 256 * 2);
  unsigned short* xc1   = (unsigned short*)alloc(256 * 256 * 2);
  unsigned short* xc2   = (unsigned short*)alloc(256 * 256 * 2);
  unsigned short* clsin = (unsigned short*)alloc(256 * 768 * 2);
  unsigned short* hcls  = (unsigned short*)alloc(256 * 768 * 2);

  // ---- GNN (shared params) on each graph
  for (int g = 0; g < 3; ++g) {
    const int Nn = gN[g], Ne = gE[g];
    // embed: relu(X @ We + be), K=200 zero-padded inside kernel
    launch_gemm<0,false,true,0>(stream, gx[g],200,200, nullptr,0,0, nullptr,
        embWbf,256, embB, Nn,256, 1,nullptr, nullptr,0, x_cur,256, nullptr,nullptr,0);
    for (int l = 0; l < 3; ++l) {
      (void)hipMemsetAsync(agg, 0, (size_t)Nn * 256 * 4, stream);
      // msg: relu(concat(x[src], edge_attr) @ Wm + bm) with fused atomic segment_sum
      launch_gemm<1,true,false,2>(stream, x_cur,256,256, gea[g],128,128, gei[g],
          msgWbf + (long)l*384*256, 256, msgB + l*256, Ne,256, 1,nullptr,
          nullptr,0, nullptr,0, agg, gei[g] + Ne, 256);
      // upd: relu(concat(x, agg) @ Wu + bu)
      launch_gemm<1,false,false,0>(stream, x_cur,256,256, agg,256,256, nullptr,
          updWbf + (long)l*512*256, 256, updB + l*256, Nn,256, 1,nullptr,
          nullptr,0, x_nxt,256, nullptr,nullptr,0);
      unsigned short* tswap = x_cur; x_cur = x_nxt; x_nxt = tswap;
    }
    (void)hipMemsetAsync(embF[g], 0, 256 * 256 * 4, stream);
    pool_kernel<<<Nn, 256, 0, stream>>>(x_cur, gfw[g], gbt[g], embF[g], Nn);
  }

  // ---- encoder stack (seq len == 1 => softmax trivial => mha = (kv@Wv+bv)@Wo+bo)
  auto runEnc = [&](const Enc& e, const EncBf& w, unsigned short* x, const unsigned short* mem)
  {
    for (int l = 0; l < 2; ++l) {
      const unsigned short* kv;
      if (mem) { kv = mem; }                       // cross: q irrelevant under len-1 softmax
      else {
        ln_kernel<<<256, 256, 0, stream>>>(x, e.ln1g + l*256, e.ln1b + l*256, hbuf);
        kv = hbuf;
      }
      // v = kv @ Wv + bv  (Wv = cols [512,768) of qkv.W)
      launch_gemm<0,false,false,0>(stream, kv,256,256, nullptr,0,0, nullptr,
          w.q + (long)l*256*768 + 512, 768, e.qkvb + l*768 + 512,
          256,256, 0,nullptr, nullptr,0, vbuf,256, nullptr,nullptr,0);
      // x = x + v @ Wo + bo
      launch_gemm<0,false,false,0>(stream, vbuf,256,256, nullptr,0,0, nullptr,
          w.o + (long)l*256*256, 256, e.outb + l*256,
          256,256, 0,nullptr, x,256, x,256, nullptr,nullptr,0);
      ln_kernel<<<256, 256, 0, stream>>>(x, e.ln2g + l*256, e.ln2b + l*256, hbuf);
      // ff1 relu
      launch_gemm<0,false,false,0>(stream, hbuf,256,256, nullptr,0,0, nullptr,
          w.f1 + (long)l*256*1024, 1024, e.ff1b + l*1024,
          256,1024, 1,nullptr, nullptr,0, fbuf,1024, nullptr,nullptr,0);
      // x = x + ff2(fbuf)
      launch_gemm<0,false,false,0>(stream, fbuf,1024,1024, nullptr,0,0, nullptr,
          w.f2 + (long)l*1024*256, 256, e.ff2b + l*256,
          256,256, 0,nullptr, x,256, x,256, nullptr,nullptr,0);
    }
  };

  auto runBranch = [&](int bi, const Enc& sa, const EncBf& saw,
                       const Enc& ca, const EncBf& caw, unsigned short* xc, float* embAdd)
  {
    // fused = prelu(concat(add_emb, main_emb) @ Wf + bf)
    launch_gemm<1,false,true,0>(stream, embAdd,256,256, embF[0],256,256, nullptr,
        fuseWbf[bi],256, fuseB[bi], 256,256, 2, fuseA[bi],
        nullptr,0, sx,256, nullptr,nullptr,0);
    runEnc(sa, saw, sx, nullptr);                   // self encoder (T=2)
    castTo(embF[0],1,256, xc,0,256, 256,256);       // cross query init = main_emb
    runEnc(ca, caw, xc, sx);                        // cross encoder (T=2)
  };
  runBranch(0, sa1, sa1b, ca1, ca1b, xc1, embF[1]);
  runBranch(1, sa2, sa2b, ca2, ca2b, xc2, embF[2]);

  // ---- classifier: cls_in = [main_emb | c1 | c2] (256 x 768)
  castTo(embF[0],1,256, clsin + 0,   0,768, 256,256);
  castTo(xc1,    0,256, clsin + 256, 0,768, 256,256);
  castTo(xc2,    0,256, clsin + 512, 0,768, 256,256);
  // h = prelu(cls_in @ W1 + b1)
  launch_gemm<0,false,false,0>(stream, clsin,768,768, nullptr,0,0, nullptr,
      clsW1bf,768, clsB1, 256,768, 2, clsA, nullptr,0, hcls,768, nullptr,nullptr,0);
  // out = sigmoid(h @ W2 + b2) -> f32 d_out
  launch_gemm<0,false,false,1>(stream, hcls,768,768, nullptr,0,0, nullptr,
      clsW2bf,1024, clsB2, 256,1024, 3,nullptr, nullptr,0, d_out,1024, nullptr,nullptr,0);
}
